// GGUFLinear_34746285425026
// MI455X (gfx1250) — matrix operand order repaired and checked by reading, not compiled, not run on hardware
//
#include <hip/hip_runtime.h>

typedef __attribute__((ext_vector_type(16))) _Float16 v16h;
typedef __attribute__((ext_vector_type(8)))  _Float16 v8h;
typedef __attribute__((ext_vector_type(4)))  _Float16 v4h;
typedef __attribute__((ext_vector_type(8)))  float    v8f;
typedef __attribute__((ext_vector_type(4)))  float    v4f;
typedef __attribute__((ext_vector_type(4)))  int      v4i;

#define TILE_M 128
#define TILE_N 256
#define TILE_K 32
#define STRIDE 40   // 32 halfs + 8 pad -> 80B rows, 16B aligned, conflict-free b128 frags

__global__ __launch_bounds__(256, 1)
void gguf_linear_wmma(const float* __restrict__ x,
                      const int*   __restrict__ wq,
                      const float* __restrict__ scales,
                      const float* __restrict__ bias,
                      float* __restrict__ out,
                      int M, int N, int K)
{
    __shared__ _Float16 ldsA[2][TILE_M * STRIDE];   // 2 x 10 KB
    __shared__ _Float16 ldsB[2][TILE_N * STRIDE];   // 2 x 20 KB

    const int tid  = threadIdx.x;
    const int lane = tid & 31;
    const int wave = tid >> 5;
    const int wm   = wave >> 2;          // 0..1 -> 64-row band
    const int wn   = wave & 3;           // 0..3 -> 64-col band

    const int m0 = blockIdx.x * TILE_M;  // M fastest-varying: concurrent blocks span
    const int n0 = blockIdx.y * TILE_N;  // all M for a narrow N window -> x stays in L2

    const int r = lane & 15;
    const int h = lane >> 4;

    // staging geometry: each thread owns column chunk (tid&7)*4, rows base+32*t
    const int srow = tid >> 3;           // 0..31
    const int scol = (tid & 7) << 2;     // 0,4,...,28

    const int kblocks = K / TILE_K;      // 128 (TILE_K == Q8_0 block)
    const int sstride = K >> 5;

    v8f acc[4][4];
#pragma unroll
    for (int i = 0; i < 4; ++i)
#pragma unroll
        for (int j = 0; j < 4; ++j)
            acc[i][j] = (v8f){0.f, 0.f, 0.f, 0.f, 0.f, 0.f, 0.f, 0.f};

    // register-staged tiles (pipeline depth 1)
    v4f  xr[4];
    v4i  wr[8];
    float ws[8];

    const float* xbase = x  + (size_t)(m0 + srow) * K + scol;
    const int*   wbase = wq + (size_t)(n0 + srow) * K + scol;

    // ---- issue global loads for tile kb (no waits here) ----
    auto issue = [&](int kb) {
        const size_t k0 = (size_t)kb * TILE_K;
#pragma unroll
        for (int t = 0; t < 4; ++t)
            xr[t] = *(const v4f*)(xbase + k0 + (size_t)t * 32 * K);
#pragma unroll
        for (int t = 0; t < 8; ++t)
            wr[t] = *(const v4i*)(wbase + k0 + (size_t)t * 32 * K);
#pragma unroll
        for (int t = 0; t < 8; ++t)
            ws[t] = scales[(size_t)(n0 + srow + 32 * t) * sstride + kb];
    };

    // ---- convert staged registers into LDS buffer `buf` ----
    auto stash = [&](int buf) {
#pragma unroll
        for (int t = 0; t < 4; ++t) {
            v4h hv;
            hv[0] = (_Float16)xr[t].x; hv[1] = (_Float16)xr[t].y;
            hv[2] = (_Float16)xr[t].z; hv[3] = (_Float16)xr[t].w;
            *(v4h*)&ldsA[buf][(srow + 32 * t) * STRIDE + scol] = hv;
        }
#pragma unroll
        for (int t = 0; t < 8; ++t) {
            const float s = ws[t];
            v4h hv;
            hv[0] = (_Float16)((float)(wr[t].x - 128) * s);
            hv[1] = (_Float16)((float)(wr[t].y - 128) * s);
            hv[2] = (_Float16)((float)(wr[t].z - 128) * s);
            hv[3] = (_Float16)((float)(wr[t].w - 128) * s);
            *(v4h*)&ldsB[buf][(srow + 32 * t) * STRIDE + scol] = hv;
        }
    };

    // prologue: fill buffer 0
    issue(0);
    stash(0);

    for (int kb = 0; kb < kblocks; ++kb) {
        const int buf = kb & 1;

        // staging regs are dead here (consumed by previous stash): fire next
        // tile's loads BEFORE the barrier so they also overlap barrier slack.
        if (kb + 1 < kblocks) {
            issue(kb + 1);
            __builtin_prefetch(xbase + (size_t)(kb + 2) * TILE_K, 0, 0);
            __builtin_prefetch(wbase + (size_t)(kb + 2) * TILE_K, 0, 0);
        }

        __syncthreads();                      // buf ready; buf^1 free for stash

        const _Float16* A = ldsA[buf];
        const _Float16* B = ldsB[buf];

        // A fragments: 16x32 f16 per ISA layout (lane r = row, h = K-half)
        v16h af[4];
#pragma unroll
        for (int am = 0; am < 4; ++am) {
            const int row = wm * 64 + am * 16 + r;
            const v8h lo = *(const v8h*)&A[row * STRIDE + h * 8];
            const v8h hi = *(const v8h*)&A[row * STRIDE + 16 + h * 8];
#pragma unroll
            for (int i = 0; i < 8; ++i) { af[am][i] = lo[i]; af[am][8 + i] = hi[i]; }
        }

        // B fragments: 32x16 f16 (lane r = out col, h selects K 8h..8h+7 / 16+8h..16+8h+7)
#pragma unroll
        for (int bn = 0; bn < 4; ++bn) {
            const int row = wn * 64 + bn * 16 + r;
            const v8h lo = *(const v8h*)&B[row * STRIDE + h * 8];
            const v8h hi = *(const v8h*)&B[row * STRIDE + 16 + h * 8];
            v16h bf;
#pragma unroll
            for (int i = 0; i < 8; ++i) { bf[i] = lo[i]; bf[8 + i] = hi[i]; }
#pragma unroll
            for (int am = 0; am < 4; ++am) {
                acc[am][bn] = __builtin_amdgcn_wmma_f32_16x16x32_f16(
                    false, af[am], false, bf, (short)0, acc[am][bn], false, false);
            }
        }

        // convert next tile after compute: VALU co-executes with in-flight WMMAs
        if (kb + 1 < kblocks)
            stash(buf ^ 1);
    }

    // epilogue: C layout — VGPR v: M = base + h*8 + v, N = lane&15
#pragma unroll
    for (int bn = 0; bn < 4; ++bn) {
        const int col = n0 + wn * 64 + bn * 16 + r;
        const float bv = bias[col];
#pragma unroll
        for (int am = 0; am < 4; ++am) {
            const int rowbase = m0 + wm * 64 + am * 16 + h * 8;
#pragma unroll
            for (int v = 0; v < 8; ++v) {
                out[(size_t)(rowbase + v) * N + col] = acc[am][bn][v] + bv;
            }
        }
    }
}

extern "C" void kernel_launch(void* const* d_in, const int* in_sizes, int n_in,
                              void* d_out, int out_size, void* d_ws, size_t ws_size,
                              hipStream_t stream) {
    const float* x      = (const float*)d_in[0];
    const int*   wq     = (const int*)d_in[1];
    const float* scales = (const float*)d_in[2];
    const float* bias   = (const float*)d_in[3];
    float*       out    = (float*)d_out;

    const int N = in_sizes[3];            // D_OUT = 16384
    const int K = in_sizes[1] / N;        // D_IN  = 4096
    const int M = in_sizes[0] / K;        // B*S   = 4096

    dim3 grid(M / TILE_M, N / TILE_N);    // (32, 64); M fast for L2 reuse of x
    gguf_linear_wmma<<<grid, 256, 0, stream>>>(x, wq, scales, bias, out, M, N, K);
}
